// SAModule_62947040690654
// MI455X (gfx1250) — compile-verified, matrix operand order checked by reading, not run on hardware
//
#include <hip/hip_runtime.h>
#include <hip/hip_fp16.h>
#include <stdint.h>

#define BB   8
#define NPTS 4096
#define MPTS 2048
#define KNN  32
#define RAD2 0.04f
#define DH   64
#define DO   128
#define HST  72   // padded f16 row stride for h1 (64 cols) -> 144B, 16B aligned, bank-rotating
#define WST  72   // padded f16 row stride for W2^T (64 k)   -> 144B

typedef __attribute__((ext_vector_type(8)))  _Float16 v8h;
typedef __attribute__((ext_vector_type(16))) _Float16 v16h;
typedef __attribute__((ext_vector_type(8)))  float    v8f;

typedef __attribute__((address_space(1))) int gint_t;   // global (addrspace 1)
typedef __attribute__((address_space(3))) int lint_t;   // LDS    (addrspace 3)

// -------------------------------------------------------------------------
// Kernel 1: farthest point sampling. One 1024-thread block per cloud.
// Each thread owns 4 points (coords + running min-dist in registers).
// Per step: wave32 shuffle argmax -> LDS cross-wave argmax -> owner thread
// broadcasts the new centroid's coords from registers via LDS.
// Also writes qpos output and the batch-id output section.
// -------------------------------------------------------------------------
__global__ __launch_bounds__(1024) void fps_kernel(const float* __restrict__ pos,
                                                   float* __restrict__ qout,  // [B*MP*3]
                                                   int*   __restrict__ bout)  // [B*MP]
{
  const int b    = blockIdx.x;
  const int tid  = threadIdx.x;
  const int lane = tid & 31;
  const int wave = tid >> 5;
  const float* p = pos + (size_t)b * NPTS * 3;

  __shared__ float swv[32];
  __shared__ int   swi[32];
  __shared__ int   sSel;
  __shared__ float sCx, sCy, sCz;

  float px[4], py[4], pz[4], md[4];
  const float x0 = p[0], y0 = p[1], z0 = p[2];
#pragma unroll
  for (int t = 0; t < 4; ++t) {
    const int pid = tid + t * 1024;
    px[t] = p[pid * 3 + 0];
    py[t] = p[pid * 3 + 1];
    pz[t] = p[pid * 3 + 2];
    const float dx = px[t] - x0, dy = py[t] - y0, dz = pz[t] - z0;
    md[t] = dx * dx + dy * dy + dz * dz;
  }
  if (tid == 0) {
    qout[(size_t)(b * MPTS) * 3 + 0] = x0;
    qout[(size_t)(b * MPTS) * 3 + 1] = y0;
    qout[(size_t)(b * MPTS) * 3 + 2] = z0;
  }
  for (int i = tid; i < MPTS; i += 1024) bout[b * MPTS + i] = b;

  for (int s = 1; s < MPTS; ++s) {
    // per-thread argmax over 4 owned points
    float bv = md[0]; int bidx = tid;
#pragma unroll
    for (int t = 1; t < 4; ++t)
      if (md[t] > bv) { bv = md[t]; bidx = tid + t * 1024; }
    // wave32 argmax (tie -> lowest index, matches jnp.argmax)
#pragma unroll
    for (int off = 16; off; off >>= 1) {
      const float ov = __shfl_xor(bv, off);
      const int   oi = __shfl_xor(bidx, off);
      if (ov > bv || (ov == bv && oi < bidx)) { bv = ov; bidx = oi; }
    }
    if (lane == 0) { swv[wave] = bv; swi[wave] = bidx; }
    __syncthreads();
    if (wave == 0) {
      bv = swv[lane]; bidx = swi[lane];
#pragma unroll
      for (int off = 16; off; off >>= 1) {
        const float ov = __shfl_xor(bv, off);
        const int   oi = __shfl_xor(bidx, off);
        if (ov > bv || (ov == bv && oi < bidx)) { bv = ov; bidx = oi; }
      }
      if (lane == 0) sSel = bidx;
    }
    __syncthreads();
    const int sel = sSel;
    if (tid == (sel & 1023)) {  // owning thread broadcasts from registers
      const int t = sel >> 10;
      sCx = px[t]; sCy = py[t]; sCz = pz[t];
      qout[(size_t)(b * MPTS + s) * 3 + 0] = px[t];
      qout[(size_t)(b * MPTS + s) * 3 + 1] = py[t];
      qout[(size_t)(b * MPTS + s) * 3 + 2] = pz[t];
    }
    __syncthreads();
    const float cx = sCx, cy = sCy, cz = sCz;
#pragma unroll
    for (int t = 0; t < 4; ++t) {
      const float dx = px[t] - cx, dy = py[t] - cy, dz = pz[t] - cz;
      md[t] = fminf(md[t], dx * dx + dy * dy + dz * dz);
    }
  }
}

// -------------------------------------------------------------------------
// Kernel 2: K-nearest neighbors per centroid. 256 threads/block, one thread
// per centroid; the whole cloud's positions (48KB) staged in LDS via the
// gfx1250 async-to-LDS path (builtin params are int* per the compiler's own
// diagnostic). Register-resident top-32 (fully unrolled -> no scratch).
// -------------------------------------------------------------------------
__global__ __launch_bounds__(256) void knn_kernel(const float* __restrict__ pos,
                                                  const float* __restrict__ qout,
                                                  int* __restrict__ nbr)  // [B*MP*K]
{
  __shared__ __align__(16) float spx[NPTS];
  __shared__ __align__(16) float spy[NPTS];
  __shared__ __align__(16) float spz[NPTS];

  const int tid  = threadIdx.x;
  const int cent = blockIdx.x * 256 + tid;     // 2048 % 256 == 0 -> block is cloud-pure
  const int b    = cent / MPTS;
  const float* p = pos + (size_t)b * NPTS * 3;

#if defined(__gfx1250__) && __has_builtin(__builtin_amdgcn_global_load_async_to_lds_b32)
  for (int i = tid; i < NPTS; i += 256) {
    gint_t* gp = (gint_t*)(p + (size_t)i * 3);
    __builtin_amdgcn_global_load_async_to_lds_b32(gp + 0, (lint_t*)&spx[i], 0, 0);
    __builtin_amdgcn_global_load_async_to_lds_b32(gp + 1, (lint_t*)&spy[i], 0, 0);
    __builtin_amdgcn_global_load_async_to_lds_b32(gp + 2, (lint_t*)&spz[i], 0, 0);
  }
#if __has_builtin(__builtin_amdgcn_s_wait_asynccnt)
  __builtin_amdgcn_s_wait_asynccnt(0);
#else
  asm volatile("s_wait_asynccnt 0" ::: "memory");
#endif
#else
  for (int i = tid; i < NPTS; i += 256) {
    spx[i] = p[i * 3 + 0];
    spy[i] = p[i * 3 + 1];
    spz[i] = p[i * 3 + 2];
  }
#endif
  __syncthreads();

  const float qx = qout[cent * 3 + 0];
  const float qy = qout[cent * 3 + 1];
  const float qz = qout[cent * 3 + 2];

  float bd[KNN]; int bi[KNN];
#pragma unroll
  for (int s = 0; s < KNN; ++s) { bd[s] = 3.4e38f; bi[s] = 0; }
  float worst = 3.4e38f; int wslot = 0;

  for (int j = 0; j < NPTS; ++j) {
    const float dx = qx - spx[j], dy = qy - spy[j], dz = qz - spz[j];
    const float d2 = dx * dx + dy * dy + dz * dz;
    if (d2 < worst) {
#pragma unroll
      for (int s = 0; s < KNN; ++s)
        if (s == wslot) { bd[s] = d2; bi[s] = j; }
      worst = -1.0f;
#pragma unroll
      for (int s = 0; s < KNN; ++s)
        if (bd[s] > worst) { worst = bd[s]; wslot = s; }
    }
  }
#pragma unroll
  for (int s = 0; s < KNN; ++s) nbr[(size_t)cent * KNN + s] = bi[s];
}

// -------------------------------------------------------------------------
// Kernel 3: gather + MLP + masked max, WMMA for the 64->128 layer.
// Block = 256 threads = 8 waves; each wave processes 8 centroids.
// h1 (32x64, relu'd) stored as f16 in LDS with padded stride so every WMMA
// A fragment is two contiguous ds_load_b128s matching the documented 16-bit
// A layout (lane half -> K+8, VGPR group -> K+16). W2 staged transposed in
// LDS for the same reason on B fragments.
// -------------------------------------------------------------------------
__device__ inline v16h ld_frag(const _Float16* base) {
  const v8h lo = *(const v8h*)(base);        // K = k0 .. k0+7
  const v8h hi = *(const v8h*)(base + 16);   // K = k0+16 .. k0+23
  v16h r;
#pragma unroll
  for (int i = 0; i < 8; ++i) { r[i] = lo[i]; r[i + 8] = hi[i]; }
  return r;
}

__global__ __launch_bounds__(256) void mlp_kernel(const float* __restrict__ x,
                                                  const float* __restrict__ pos,
                                                  const float* __restrict__ W1,
                                                  const float* __restrict__ b1,
                                                  const float* __restrict__ W2,
                                                  const float* __restrict__ b2,
                                                  const float* __restrict__ qout,
                                                  const int* __restrict__ nbr,
                                                  float* __restrict__ out)
{
  __shared__ __align__(16) _Float16 sW2t[DO * WST];       // W2^T, f16
  __shared__ __align__(16) _Float16 sH1[8][KNN * HST];    // per-wave h1 tile
  __shared__ float sW1[6 * DH];
  __shared__ float sB1[DH];
  __shared__ float sB2[DO];

  const int tid  = threadIdx.x;
  const int lane = tid & 31;
  const int wave = tid >> 5;
  const int half = lane >> 4;
  const int m    = lane & 15;

  for (int i = tid; i < DH * DO; i += 256) {
    const int k = i / DO, n = i % DO;
    sW2t[n * WST + k] = (_Float16)W2[i];
  }
  for (int i = tid; i < 6 * DH; i += 256) sW1[i] = W1[i];
  if (tid < DH) sB1[tid] = b1[tid];
  if (tid < DO) sB2[tid] = b2[tid];
  __syncthreads();

  _Float16* h1w = sH1[wave];

  for (int cc = 0; cc < 8; ++cc) {
    const int cent = blockIdx.x * 64 + wave * 8 + cc;
    const int b    = cent / MPTS;
    const int nb   = nbr[(size_t)cent * KNN + lane];
    const float* pj = pos + (size_t)(b * NPTS + nb) * 3;
    const float* xj = x   + (size_t)(b * NPTS + nb) * 3;
    const float q0 = qout[cent * 3 + 0];
    const float q1 = qout[cent * 3 + 1];
    const float q2 = qout[cent * 3 + 2];
    const float f0 = xj[0], f1 = xj[1], f2 = xj[2];
    const float f3 = pj[0] - q0, f4 = pj[1] - q1, f5 = pj[2] - q2;
    const float d2 = f3 * f3 + f4 * f4 + f5 * f5;
    const bool valid = d2 <= RAD2;
    const unsigned mask = (unsigned)__ballot(valid);   // bit k = neighbor k valid

    // layer 1: lane = neighbor row, 6-deep FMA chain, relu, f16 to LDS
#pragma unroll 8
    for (int c = 0; c < DH; ++c) {
      float a = sB1[c] + f0 * sW1[0 * DH + c] + f1 * sW1[1 * DH + c]
                       + f2 * sW1[2 * DH + c] + f3 * sW1[3 * DH + c]
                       + f4 * sW1[4 * DH + c] + f5 * sW1[5 * DH + c];
      a = fmaxf(a, 0.0f);
      h1w[lane * HST + c] = (_Float16)a;
    }
    __syncthreads();

    float cm[8];
#pragma unroll
    for (int ni = 0; ni < 8; ++ni) cm[ni] = -__builtin_inff();

#pragma unroll
    for (int mi = 0; mi < 2; ++mi) {
      const int row = 16 * mi + m;
      const v16h a0 = ld_frag(&h1w[row * HST + 8 * half]);        // K 0..31
      const v16h a1 = ld_frag(&h1w[row * HST + 32 + 8 * half]);   // K 32..63
      const unsigned rmask = (mask >> (16 * mi + 8 * half)) & 0xFFu;
#pragma unroll
      for (int ni = 0; ni < 8; ++ni) {
        const int n = ni * 16 + m;
        const v16h b0f = ld_frag(&sW2t[n * WST + 8 * half]);
        const v16h b1f = ld_frag(&sW2t[n * WST + 32 + 8 * half]);
        const float bias = sB2[n];
        v8f acc;
#pragma unroll
        for (int r = 0; r < 8; ++r) acc[r] = bias;
        acc = __builtin_amdgcn_wmma_f32_16x16x32_f16(false, a0, false, b0f,
                                                     (short)0, acc, false, false);
        acc = __builtin_amdgcn_wmma_f32_16x16x32_f16(false, a1, false, b1f,
                                                     (short)0, acc, false, false);
        // relu + neighbor-mask + max over this lane's 8 rows of the D tile
        float mx = -__builtin_inff();
#pragma unroll
        for (int r = 0; r < 8; ++r) {
          const float h = fmaxf(acc[r], 0.0f);
          mx = fmaxf(mx, ((rmask >> r) & 1u) ? h : -__builtin_inff());
        }
        const float omx = __shfl_xor(mx, 16);   // other half: rows +/-8
        mx = fmaxf(mx, omx);
        cm[ni] = fmaxf(cm[ni], mx);
      }
    }
    if (lane < 16) {
#pragma unroll
      for (int ni = 0; ni < 8; ++ni)
        out[(size_t)cent * DO + ni * 16 + lane] = fmaxf(cm[ni], 0.0f); // -inf -> 0
    }
    __syncthreads();
  }
}

// -------------------------------------------------------------------------
extern "C" void kernel_launch(void* const* d_in, const int* in_sizes, int n_in,
                              void* d_out, int out_size, void* d_ws, size_t ws_size,
                              hipStream_t stream) {
  const float* x   = (const float*)d_in[0];
  const float* pos = (const float*)d_in[1];
  /* d_in[2] = batch (unused: batch id per centroid == cloud index) */
  const float* W1  = (const float*)d_in[3];
  const float* b1  = (const float*)d_in[4];
  const float* W2  = (const float*)d_in[5];
  const float* b2  = (const float*)d_in[6];

  float* out  = (float*)d_out;                          // [B*MP, 128]
  float* qout = out + (size_t)BB * MPTS * DO;           // [B*MP, 3]
  int*   bout = (int*)(qout + (size_t)BB * MPTS * 3);   // [B*MP] int32 bits
  int*   nbr  = (int*)d_ws;                             // [B*MP, K] = 2MB

  fps_kernel<<<BB, 1024, 0, stream>>>(pos, qout, bout);
  knn_kernel<<<(BB * MPTS) / 256, 256, 0, stream>>>(pos, qout, nbr);
  mlp_kernel<<<(BB * MPTS) / 64, 256, 0, stream>>>(x, pos, W1, b1, W2, b2, qout, nbr, out);
}
